// QINCoStep_25348896981484
// MI455X (gfx1250) — matrix operand order repaired and checked by reading, not compiled, not run on hardware
//
#include <hip/hip_runtime.h>

typedef _Float16 v16h __attribute__((ext_vector_type(16)));
typedef _Float16 v8h  __attribute__((ext_vector_type(8)));
typedef _Float16 v4h  __attribute__((ext_vector_type(4)));
typedef float    v8f  __attribute__((ext_vector_type(8)));

#define BSZ 2048
#define DD  128
#define KK  256
#define HH  256
#define ZS  136   // padded half-stride for 128-wide rows (+16B) -> conflict-free b128 LDS reads
#define HS  40    // padded half-stride for per-wave h scratch (32 rows: both M-tiles)

#define WMMA_F16(A,B,C) \
  __builtin_amdgcn_wmma_f32_16x16x32_f16(false,(A),false,(B),(short)0,(C),false,false)

// single-instruction relu: v_med3_f32(x, 0, BIG) -- no canonicalize pair
__device__ __forceinline__ float relu1(float x){
  return __builtin_amdgcn_fmed3f(x, 0.f, 3.0e38f);
}

// A-fragment (16x32 f16): lane holds K = kb..kb+7 and 16+kb..16+kb+7 (kb = 0 or 8)
__device__ __forceinline__ v16h frag_split16(const _Float16* p){
  v8h lo = *(const v8h*)p;
  v8h hi = *(const v8h*)(p + 16);
  return __builtin_shufflevector(lo, hi, 0,1,2,3,4,5,6,7,8,9,10,11,12,13,14,15);
}
// B-fragment (32x16 f16): lane holds K = kb..kb+7 and 16+kb..16+kb+7 of one column n (= row of W)
__device__ __forceinline__ v16h frag_contig(const _Float16* p){
  v8h lo = *(const v8h*)p;
  v8h hi = *(const v8h*)(p + 16);
  return __builtin_shufflevector(lo, hi, 0,1,2,3,4,5,6,7,8,9,10,11,12,13,14,15);
}

// One residual MLP block; each wave owns M-tiles (wave) and (wave+8), processed
// together: every W1/W2 B-fragment is loaded once and feeds two WMMAs. Loads are
// phase-separated from the WMMA bursts so address VALU / DS issue overlaps the
// previous burst instead of creating WMMA->VALU WAR hazard NOPs.
__device__ __forceinline__ void mlp_block(_Float16* sZ, const _Float16* sW1,
                                          _Float16* hS,
                                          const _Float16* __restrict__ W2h,
                                          int wave, int lane)
{
  const int m  = lane & 15;
  const int kb = (lane < 16) ? 0 : 8;    // A-fragment K base
  const int ks = (lane < 16) ? 0 : 8;    // B-fragment K base
  const int mr = (lane < 16) ? 0 : 8;    // C/D row base
  const int k0 = wave * 16;              // tile 0 rows
  const int k1 = (wave + 8) * 16;        // tile 1 rows

  v16h Az0[4], Az1[4];
#pragma unroll
  for (int kc = 0; kc < 4; ++kc){
    Az0[kc] = frag_split16(&sZ[(k0 + m)*ZS + kc*32 + kb]);
    Az1[kc] = frag_split16(&sZ[(k1 + m)*ZS + kc*32 + kb]);
  }

  v8f out0[8] = {}, out1[8] = {};        // 2 x (16x128) f32 accumulators

  for (int hc = 0; hc < 8; ++hc){        // stream h in 32-column chunks
    const int n0 = hc * 32;

    // ---- layer 1: load all B fragments, then 16-WMMA burst ----
    v16h B0[4], B1[4];
#pragma unroll
    for (int kc = 0; kc < 4; ++kc){
      B0[kc] = frag_contig(&sW1[(n0      + m)*ZS + kc*32 + ks]);
      B1[kc] = frag_contig(&sW1[(n0 + 16 + m)*ZS + kc*32 + ks]);
    }
    v8f h00 = {}, h01 = {}, h10 = {}, h11 = {};
#pragma unroll
    for (int kc = 0; kc < 4; ++kc){
      h00 = WMMA_F16(Az0[kc], B0[kc], h00);
      h10 = WMMA_F16(Az1[kc], B0[kc], h10);
      h01 = WMMA_F16(Az0[kc], B1[kc], h01);
      h11 = WMMA_F16(Az1[kc], B1[kc], h11);
    }

    // ---- layer 2: prefetch all W2 fragments (global, L2-hot) ----
    v16h B2[8];
#pragma unroll
    for (int nt = 0; nt < 8; ++nt)
      B2[nt] = frag_contig(&W2h[(nt*16 + m)*HH + n0 + ks]);

    // relu -> f16 -> per-wave LDS scratch (D-layout -> A-layout transpose)
#pragma unroll
    for (int j = 0; j < 8; ++j){
      hS[(mr + j)*HS + m]           = (_Float16)relu1(h00[j]);
      hS[(mr + j)*HS + 16 + m]      = (_Float16)relu1(h01[j]);
      hS[(16 + mr + j)*HS + m]      = (_Float16)relu1(h10[j]);
      hS[(16 + mr + j)*HS + 16 + m] = (_Float16)relu1(h11[j]);
    }
    v16h A20 = frag_split16(&hS[m*HS + kb]);        // same-wave DS ops in order
    v16h A21 = frag_split16(&hS[(16 + m)*HS + kb]);

    // ---- layer 2: 16-WMMA burst ----
#pragma unroll
    for (int nt = 0; nt < 8; ++nt){
      out0[nt] = WMMA_F16(A20, B2[nt], out0[nt]);
      out1[nt] = WMMA_F16(A21, B2[nt], out1[nt]);
    }
  }
  // residual: z <- z + out (read-modify-write in D layout)
#pragma unroll
  for (int nt = 0; nt < 8; ++nt)
#pragma unroll
    for (int j = 0; j < 8; ++j){
      const int c  = nt*16 + m;
      const int r0 = k0 + mr + j;
      const int r1 = k1 + mr + j;
      sZ[r0*ZS + c] = (_Float16)((float)sZ[r0*ZS + c] + out0[nt][j]);
      sZ[r1*ZS + c] = (_Float16)((float)sZ[r1*ZS + c] + out1[nt][j]);
    }
}

// --- prep kernels -----------------------------------------------------------

__global__ void cvt_w_f16(const float* __restrict__ s, _Float16* __restrict__ d){
  const int i = blockIdx.x * 256 + threadIdx.x;     // exactly 32768 threads
  d[i] = (_Float16)s[i];
}

// base[k,dd] = codebook[k,dd] + sum_j codebook[k,j]*Wc[j,dd] + bc[dd]
__global__ void prep_base(const float* __restrict__ cb, const float* __restrict__ Wc,
                          const float* __restrict__ bc, float* __restrict__ base){
  const int k = blockIdx.x, dd = threadIdx.x;       // 128 threads
  __shared__ float row[DD];
  row[dd] = cb[k*DD + dd];
  __syncthreads();
  float acc = bc[dd] + row[dd];
  for (int j = 0; j < DD; ++j) acc += row[j] * Wc[j*(2*DD) + dd];
  base[k*DD + dd] = acc;
}

// proj[b,dd] = sum_j xhat[b,j]*Wc[j,128+dd]
__global__ void prep_proj(const float* __restrict__ xhat, const float* __restrict__ Wc,
                          float* __restrict__ proj){
  const int b = blockIdx.x, dd = threadIdx.x;       // 128 threads
  __shared__ float row[DD];
  row[dd] = xhat[b*DD + dd];
  __syncthreads();
  float acc = 0.f;
  for (int j = 0; j < DD; ++j) acc += row[j] * Wc[j*(2*DD) + DD + dd];
  proj[b*DD + dd] = acc;
}

// --- main fused kernel: one workgroup (8 wave32) per batch row --------------

__global__ __launch_bounds__(256, 1)
void qinco_main(const float* __restrict__ basep, const float* __restrict__ projp,
                const _Float16* __restrict__ W1a, const _Float16* __restrict__ W2a,
                const _Float16* __restrict__ W1b, const _Float16* __restrict__ W2b,
                const float* __restrict__ xhat, const float* __restrict__ x,
                int* __restrict__ codes, float* __restrict__ delta)
{
  __shared__ _Float16 sZ [KK * ZS];       // 69,632 B  (z, f16, padded)
  __shared__ _Float16 sW1[HH * ZS];       // 69,632 B  (current block's W1, f16, padded)
  __shared__ _Float16 sH [8][32 * HS];    // 20,480 B  (per-wave h chunk scratch, 2 tiles)
  __shared__ float    sDiff[DD];
  __shared__ float    sDist[KK];
  __shared__ int      sIdx [KK];

  const int b    = blockIdx.x;
  const int tid  = threadIdx.x;
  const int wave = tid >> 5, lane = tid & 31;
  const float* projb = projp + b * DD;

  // stage z = base + proj (f32+f32 -> f16 LDS)
  for (int i = tid; i < (KK*DD)/4; i += 256){
    const int idx = i*4, r = idx >> 7, c = idx & 127;
    float4 bv = ((const float4*)basep)[i];
    float4 pv = ((const float4*)projb)[c >> 2];
    v4h h;
    h[0] = (_Float16)(bv.x + pv.x); h[1] = (_Float16)(bv.y + pv.y);
    h[2] = (_Float16)(bv.z + pv.z); h[3] = (_Float16)(bv.w + pv.w);
    *(v4h*)&sZ[r*ZS + c] = h;
  }
  // stage W1 (block 0), already f16 in workspace
  for (int i = tid; i < (HH*DD)/8; i += 256){
    const int idx = i*8, r = idx >> 7, c = idx & 127;
    *(v8h*)&sW1[r*ZS + c] = ((const v8h*)W1a)[i];
  }
  if (tid < DD) sDiff[tid] = xhat[b*DD + tid] - x[b*DD + tid];
  __syncthreads();

  mlp_block(sZ, sW1, &sH[wave][0], W2a, wave, lane);   // block 0
  __syncthreads();

  for (int i = tid; i < (HH*DD)/8; i += 256){          // restage W1 (block 1)
    const int idx = i*8, r = idx >> 7, c = idx & 127;
    *(v8h*)&sW1[r*ZS + c] = ((const v8h*)W1b)[i];
  }
  __syncthreads();

  mlp_block(sZ, sW1, &sH[wave][0], W2b, wave, lane);   // block 1
  __syncthreads();

  // epilogue: dist[k] = || z[k] + (xhat - x) ||^2 ; cand - xhat == z
  {
    float acc = 0.f;
    const _Float16* zr = &sZ[tid * ZS];
#pragma unroll
    for (int cc = 0; cc < DD/8; ++cc){              // 16 x ds_load_b128
      v8h zv = *(const v8h*)(zr + cc*8);
#pragma unroll
      for (int j = 0; j < 8; ++j){
        float v = (float)zv[j] + sDiff[cc*8 + j];
        acc += v * v;
      }
    }
    sDist[tid] = acc; sIdx[tid] = tid;
  }
  __syncthreads();
  for (int s = 128; s > 0; s >>= 1){
    if (tid < s){
      const float d1 = sDist[tid], d2 = sDist[tid + s];
      const int   i1 = sIdx[tid],  i2 = sIdx[tid + s];
      if (d2 < d1 || (d2 == d1 && i2 < i1)){ sDist[tid] = d2; sIdx[tid] = i2; }
    }
    __syncthreads();
  }
  const int best = sIdx[0];
  if (tid == 0) codes[b] = best;
  if (tid < DD) delta[b*DD + tid] = (float)sZ[best*ZS + tid];
}

// --- launcher ---------------------------------------------------------------

extern "C" void kernel_launch(void* const* d_in, const int* in_sizes, int n_in,
                              void* d_out, int out_size, void* d_ws, size_t ws_size,
                              hipStream_t stream)
{
  (void)in_sizes; (void)n_in; (void)out_size; (void)ws_size;
  const float* xhat = (const float*)d_in[0];
  const float* x    = (const float*)d_in[1];
  const float* cb   = (const float*)d_in[2];
  const float* Wc   = (const float*)d_in[3];
  const float* bc   = (const float*)d_in[4];
  const float* W1_0 = (const float*)d_in[5];
  const float* W2_0 = (const float*)d_in[6];
  const float* W1_1 = (const float*)d_in[7];
  const float* W2_1 = (const float*)d_in[8];

  char* ws = (char*)d_ws;
  _Float16* w16   = (_Float16*)ws;                       // 4 * 65536 B
  float*    basep = (float*)(ws + 262144);               // 131072 B
  float*    projp = (float*)(ws + 262144 + 131072);      // 1 MB

  cvt_w_f16<<<128, 256, 0, stream>>>(W1_0, w16 + 0*32768);
  cvt_w_f16<<<128, 256, 0, stream>>>(W2_0, w16 + 1*32768);
  cvt_w_f16<<<128, 256, 0, stream>>>(W1_1, w16 + 2*32768);
  cvt_w_f16<<<128, 256, 0, stream>>>(W2_1, w16 + 3*32768);
  prep_base<<<KK,  DD, 0, stream>>>(cb, Wc, bc, basep);
  prep_proj<<<BSZ, DD, 0, stream>>>(xhat, Wc, projp);

  int*   codes = (int*)d_out;
  float* delta = (float*)d_out + BSZ;
  qinco_main<<<BSZ, 256, 0, stream>>>(basep, projp,
      w16 + 0*32768, w16 + 1*32768, w16 + 2*32768, w16 + 3*32768,
      xhat, x, codes, delta);
}
